// StageBlock_73693048865374
// MI455X (gfx1250) — matrix operand order repaired and checked by reading, not compiled, not run on hardware
//
#include <hip/hip_runtime.h>
#include <cmath>

typedef __attribute__((ext_vector_type(16))) _Float16 v16h;
typedef __attribute__((ext_vector_type(8)))  _Float16 v8h;
typedef __attribute__((ext_vector_type(8)))  float    v8f;

union ABu { v16h v; v8h h[2]; };

#define CH 32
#define TILE_W 128

// ------------------------------------------------------------------
// L = grad_T(grad(v)) composed stencil (matches reference boundary
// handling: gx[0]=0, gy[0,:]=0, grad_T drops first col/row then pads)
// ------------------------------------------------------------------
__device__ __forceinline__ float lap_op(const float* __restrict__ v,
                                        int h, int wv, int H, int W) {
  const float c = v[(size_t)h * W + wv];
  float Xw  = (wv >= 1)       ? c - v[(size_t)h * W + wv - 1]       : 0.f;
  float Xw1 = (wv + 1 <= W-1) ? v[(size_t)h * W + wv + 1] - c       : 0.f;
  float Yh  = (h  >= 1)       ? c - v[(size_t)(h-1) * W + wv]       : 0.f;
  float Yh1 = (h + 1 <= H-1)  ? v[(size_t)(h+1) * W + wv] - c       : 0.f;
  return Xw - Xw1 + Yh - Yh1;
}

// c = grad_T(grad_y - a); also init vk = phik = x
__global__ __launch_bounds__(256) void compute_c_init_kernel(
    const float* __restrict__ x, const float* __restrict__ grad_y,
    const float* __restrict__ a, float* __restrict__ cbuf,
    float* __restrict__ vk, float* __restrict__ phik,
    int B, int H, int W) {
  int idx = blockIdx.x * 256 + threadIdx.x;
  if (idx >= B * H * W) return;
  int wv = idx % W; int h = (idx / W) % H; int bi = idx / (W * H);
  auto G = [&](int hh, int ww, int k) -> float {
    size_t o = ((((size_t)bi) * H + hh) * W + ww) * 2 + k;
    return grad_y[o] - a[o];
  };
  float Xw  = (wv >= 1)       ? G(h, wv,   0) : 0.f;
  float Xw1 = (wv + 1 <= W-1) ? G(h, wv+1, 0) : 0.f;
  float Yh  = (h  >= 1)       ? G(h,   wv, 1) : 0.f;
  float Yh1 = (h + 1 <= H-1)  ? G(h+1, wv, 1) : 0.f;
  cbuf[idx] = Xw - Xw1 + Yh - Yh1;
  float xv = x[idx];
  vk[idx] = xv; phik[idx] = xv;
}

// one FISTA iteration; host passes coef = (t-1)/t_next
__global__ __launch_bounds__(256) void fista_step_kernel(
    const float* __restrict__ vk, const float* __restrict__ phik,
    const float* __restrict__ cbuf, const float* __restrict__ lba,
    float coef, float* __restrict__ vk_out, float* __restrict__ phik_out,
    int B, int H, int W) {
  int idx = blockIdx.x * 256 + threadIdx.x;
  if (idx >= B * H * W) return;
  int wv = idx % W; int h = (idx / W) % H; int bi = idx / (W * H);
  const float* v = vk + (size_t)bi * H * W;
  float L = lap_op(v, h, wv, H, W);
  float phinext = v[(size_t)h * W + wv] - lba[bi] * (L - cbuf[idx]);
  vk_out[idx]   = phinext + coef * (phinext - phik[idx]);
  phik_out[idx] = phinext;
}

// x_cat into padded NHWC f16 buffer: ch0 = xk, ch1 = L(xk) - c
__global__ __launch_bounds__(256) void build_xcat_kernel(
    const float* __restrict__ phik, const float* __restrict__ cbuf,
    _Float16* __restrict__ act, int B, int H, int W) {
  int idx = blockIdx.x * 256 + threadIdx.x;
  if (idx >= B * H * W) return;
  int wv = idx % W; int h = (idx / W) % H; int bi = idx / (W * H);
  const float* v = phik + (size_t)bi * H * W;
  float L = lap_op(v, h, wv, H, W);
  size_t pb = ((((size_t)bi) * (H + 2) + (h + 1)) * (W + 2) + (wv + 1)) * CH;
  v8h z = {}; v8h v0 = {};
  v0[0] = (_Float16)v[(size_t)h * W + wv];
  v0[1] = (_Float16)(L - cbuf[idx]);
  v8h* dst = (v8h*)&act[pb];
  dst[0] = v0; dst[1] = z; dst[2] = z; dst[3] = z;
}

// xk = phik*w + (1-w)*subnet(ch0); write d_out[0:BHW] and xkbuf
__global__ __launch_bounds__(256) void blend_kernel(
    const float* __restrict__ phik, const _Float16* __restrict__ act,
    const float* __restrict__ wparam, float* __restrict__ xkbuf,
    float* __restrict__ out_xk, int B, int H, int W) {
  int idx = blockIdx.x * 256 + threadIdx.x;
  if (idx >= B * H * W) return;
  int wv = idx % W; int h = (idx / W) % H; int bi = idx / (W * H);
  size_t pb = ((((size_t)bi) * (H + 2) + (h + 1)) * (W + 2) + (wv + 1)) * CH;
  float net = (float)act[pb];
  float wb = wparam[bi];
  float xk = phik[idx] * wb + (1.f - wb) * net;
  xkbuf[idx] = xk; out_xk[idx] = xk;
}

// b = grad_y - grad(xk) -> padded NHWC ch0/ch1 (NCHW reshape: ch0=gx comp, ch1=gy comp)
__global__ __launch_bounds__(256) void build_b_kernel(
    const float* __restrict__ xk, const float* __restrict__ grad_y,
    _Float16* __restrict__ act, int B, int H, int W) {
  int idx = blockIdx.x * 256 + threadIdx.x;
  if (idx >= B * H * W) return;
  int wv = idx % W; int h = (idx / W) % H; int bi = idx / (W * H);
  const float* v = xk + (size_t)bi * H * W;
  float c = v[(size_t)h * W + wv];
  float gx = (wv >= 1) ? c - v[(size_t)h * W + wv - 1] : 0.f;
  float gy = (h  >= 1) ? c - v[(size_t)(h-1) * W + wv] : 0.f;
  size_t go = (size_t)idx * 2;
  size_t pb = ((((size_t)bi) * (H + 2) + (h + 1)) * (W + 2) + (wv + 1)) * CH;
  v8h z = {}; v8h v0 = {};
  v0[0] = (_Float16)(grad_y[go + 0] - gx);
  v0[1] = (_Float16)(grad_y[go + 1] - gy);
  v8h* dst = (v8h*)&act[pb];
  dst[0] = v0; dst[1] = z; dst[2] = z; dst[3] = z;
}

// soft threshold of stblock output ch0/ch1 -> d_out[BHW:], layout (B,1,H,W,2)
__global__ __launch_bounds__(256) void soft_thresh_kernel(
    const _Float16* __restrict__ act, const float* __restrict__ dthr,
    float* __restrict__ out, int B, int H, int W) {
  int idx = blockIdx.x * 256 + threadIdx.x;
  if (idx >= B * H * W) return;
  int wv = idx % W; int h = (idx / W) % H; int bi = idx / (W * H);
  size_t pb = ((((size_t)bi) * (H + 2) + (h + 1)) * (W + 2) + (wv + 1)) * CH;
  float ts = dthr[bi];
  #pragma unroll
  for (int k = 0; k < 2; ++k) {
    float x2 = (float)act[pb + k];
    float r = (fabsf(x2) > ts) ? (x2 - copysignf(ts, x2)) : 0.f;
    out[(size_t)idx * 2 + k] = r;
  }
}

__global__ __launch_bounds__(256) void zero_kernel(float4* __restrict__ p, size_t n16) {
  size_t i = (size_t)blockIdx.x * 256 + threadIdx.x;
  if (i < n16) p[i] = make_float4(0.f, 0.f, 0.f, 0.f);
}

// pack f32 conv weights (OC,IC,3,3) -> f16 fragments [tap 0..8][n 0..31][k 0..31],
// zero-padded to 32x32; bias zero-padded to 32 floats
__global__ __launch_bounds__(256) void pack_weights_kernel(
    const float* __restrict__ Wsrc, const float* __restrict__ bsrc,
    _Float16* __restrict__ dst, float* __restrict__ bdst, int OC, int IC) {
  int idx = blockIdx.x * 256 + threadIdx.x;
  if (idx < 9 * 32 * 32) {
    int k = idx & 31; int n = (idx >> 5) & 31; int t = idx >> 10;
    int ky = t / 3, kx = t % 3;
    float val = (n < OC && k < IC)
              ? Wsrc[((((size_t)n) * IC + k) * 3 + ky) * 3 + kx] : 0.f;
    dst[idx] = (_Float16)val;
  }
  if (idx < 32) bdst[idx] = (idx < OC) ? bsrc[idx] : 0.f;
}

// ------------------------------------------------------------------
// conv3x3, C=32 -> C=32 via WMMA f32_16x16x32_f16 (implicit GEMM).
// Per wave: 16 pixels (M) x 32 out-channels (two N-tiles), K=32 in-
// channels -> 18 accumulating WMMAs; the A fragment (activations) is
// loaded once per tap and feeds both N-tiles, doubling WMMA per
// ds_load. Block = 8 waves = 128 pixels; LDS halo 3 x 130 x 32 f16.
// Activations live in zero-padded NHWC f16 buffers (borders stay 0).
// ------------------------------------------------------------------
__global__ __launch_bounds__(256) void conv3x3_wmma_kernel(
    const _Float16* __restrict__ act_in, _Float16* __restrict__ act_out,
    const _Float16* __restrict__ wpk, const float* __restrict__ bias,
    const float* __restrict__ alpha_p, int H, int W) {
  __shared__ _Float16 lds[3][TILE_W + 2][CH];

  const int w0 = blockIdx.x * TILE_W;
  const int h  = blockIdx.y;
  const int bi = blockIdx.z;
  const size_t planeW = (size_t)(W + 2);
  const _Float16* inb = act_in + ((size_t)bi * (H + 2)) * planeW * CH;

  // pull packed weights (18KB) toward the caches before the tap loop
  __builtin_prefetch(wpk, 0, 0);

  // cooperative halo load: 3 rows x 130 cols x 32ch (64B per position)
  const int tid = threadIdx.x;
  for (int p = tid; p < 3 * (TILE_W + 2); p += 256) {
    int r = p / (TILE_W + 2), c = p % (TILE_W + 2);
    const v8h* src = (const v8h*)(inb + (((size_t)(h + r)) * planeW + (w0 + c)) * CH);
    v8h* dl = (v8h*)&lds[r][c][0];
    dl[0] = src[0]; dl[1] = src[1]; dl[2] = src[2]; dl[3] = src[3];
  }
  __syncthreads();

  const int wave  = tid >> 5;       // 8 pixel-tiles of 16
  const int lane  = tid & 31;
  const int lrow  = lane & 15;
  const bool hi   = lane >= 16;

  v8f acc0 = {};   // oc 0..15
  v8f acc1 = {};   // oc 16..31
  #pragma unroll
  for (int t = 0; t < 9; ++t) {
    const int ky = t / 3, kx = t % 3;
    // A fragment: row M = lrow (pixel), K split lo/hi per ISA layout
    const _Float16* ap = &lds[ky][wave * 16 + lrow + kx][hi ? 8 : 0];
    ABu af; af.h[0] = *(const v8h*)ap; af.h[1] = *(const v8h*)(ap + 16);
    // B fragments: col N = lrow (oc), K split lo/hi per ISA layout (same as A)
    const _Float16* bp0 = wpk + (((size_t)t * 32) + lrow) * 32 + (hi ? 8 : 0);
    ABu bf0; bf0.h[0] = *(const v8h*)bp0; bf0.h[1] = *(const v8h*)(bp0 + 16);
    const _Float16* bp1 = bp0 + 16 * 32;
    ABu bf1; bf1.h[0] = *(const v8h*)bp1; bf1.h[1] = *(const v8h*)(bp1 + 16);
    acc0 = __builtin_amdgcn_wmma_f32_16x16x32_f16(
        false, af.v, false, bf0.v, (short)0, acc0, false, false);
    acc1 = __builtin_amdgcn_wmma_f32_16x16x32_f16(
        false, af.v, false, bf1.v, (short)0, acc1, false, false);
  }

  // epilogue: D layout -> lane<16: (M=r, N=lane); lane>=16: (M=r+8, N=lane-16)
  const int oc0 = lrow, oc1 = 16 + lrow;
  const float bv0 = bias[oc0], bv1 = bias[oc1];
  const bool prelu = (alpha_p != nullptr);
  const float al = prelu ? *alpha_p : 1.0f;
  _Float16* outb = act_out + ((size_t)bi * (H + 2)) * planeW * CH;
  const size_t rowbase = ((size_t)(h + 1)) * planeW;
  #pragma unroll
  for (int m = 0; m < 8; ++m) {
    int pix = w0 + 1 + wave * 16 + (hi ? 8 + m : m);
    size_t pbase = (rowbase + pix) * CH;
    float y0 = acc0[m] + bv0;
    float y1 = acc1[m] + bv1;
    if (prelu) {
      y0 = (y0 > 0.f) ? y0 : al * y0;
      y1 = (y1 > 0.f) ? y1 : al * y1;
    }
    outb[pbase + oc0] = (_Float16)y0;
    outb[pbase + oc1] = (_Float16)y1;
  }
}

// ------------------------------------------------------------------
extern "C" void kernel_launch(void* const* d_in, const int* in_sizes, int n_in,
                              void* d_out, int out_size, void* d_ws, size_t ws_size,
                              hipStream_t stream) {
  (void)in_sizes; (void)n_in; (void)out_size; (void)ws_size;
  const int B = 2, H = 512, W = 512;
  const size_t nBHW = (size_t)B * H * W;

  char* ws = (char*)d_ws;
  size_t off = 0;
  auto carve = [&](size_t bytes) -> void* {
    off = (off + 255) & ~(size_t)255;
    void* p = ws + off; off += bytes; return p;
  };
  float* c_buf = (float*)carve(nBHW * 4);
  float* vA    = (float*)carve(nBHW * 4);
  float* phA   = (float*)carve(nBHW * 4);
  float* vB    = (float*)carve(nBHW * 4);
  float* phB   = (float*)carve(nBHW * 4);
  float* xkbuf = (float*)carve(nBHW * 4);
  const size_t actHalves = (size_t)B * (H + 2) * (W + 2) * CH;
  _Float16* actA = (_Float16*)carve(actHalves * 2);
  _Float16* actB = (_Float16*)carve(actHalves * 2);
  _Float16* wpk[13]; float* bpk[13];
  for (int i = 0; i < 13; ++i) {
    wpk[i] = (_Float16*)carve(9 * 32 * 32 * 2);
    bpk[i] = (float*)carve(32 * 4);
  }

  const float* x    = (const float*)d_in[0];
  const float* a    = (const float*)d_in[1];
  const float* gy   = (const float*)d_in[2];
  const float* lba  = (const float*)d_in[3];
  const float* wprm = (const float*)d_in[4];
  const float* dthr = (const float*)d_in[5];

  // zero both padded activation buffers (contiguous carve region)
  {
    size_t n16 = (actHalves * 2 * 2) / 16;
    zero_kernel<<<dim3((unsigned)((n16 + 255) / 256)), 256, 0, stream>>>((float4*)actA, n16);
  }

  // pack weights: xblock convs at d_in[6..19], stblock convs at d_in[26..37]
  const int ocX[7] = {32,32,32,32,32,32, 1}, icX[7] = { 2,32,32,32,32,32,32};
  for (int l = 0; l < 7; ++l)
    pack_weights_kernel<<<36, 256, 0, stream>>>(
        (const float*)d_in[6 + 2*l], (const float*)d_in[7 + 2*l],
        wpk[l], bpk[l], ocX[l], icX[l]);
  const int ocS[6] = {32,32,32,32,32, 2}, icS[6] = { 2,32,32,32,32,32};
  for (int l = 0; l < 6; ++l)
    pack_weights_kernel<<<36, 256, 0, stream>>>(
        (const float*)d_in[26 + 2*l], (const float*)d_in[27 + 2*l],
        wpk[7 + l], bpk[7 + l], ocS[l], icS[l]);

  const int ew_blocks = (int)((nBHW + 255) / 256);
  compute_c_init_kernel<<<ew_blocks, 256, 0, stream>>>(x, gy, a, c_buf, vA, phA, B, H, W);

  // FISTA: 10 stencil iterations, momentum coefficients computed host-side
  double t = 1.0;
  float *vin = vA, *pin = phA, *vout = vB, *pout = phB;
  for (int i = 0; i < 10; ++i) {
    double tn = 0.5 * (1.0 + sqrt(1.0 + 4.0 * t * t));
    float coef = (float)((t - 1.0) / tn); t = tn;
    fista_step_kernel<<<ew_blocks, 256, 0, stream>>>(vin, pin, c_buf, lba, coef, vout, pout, B, H, W);
    float* tmp;
    tmp = vin; vin = vout; vout = tmp;
    tmp = pin; pin = pout; pout = tmp;
  }
  // final phik is in `pin`

  build_xcat_kernel<<<ew_blocks, 256, 0, stream>>>(pin, c_buf, actA, B, H, W);

  dim3 cgrid(W / TILE_W, H, B);
  // xblock: 7 conv layers, PReLU(alphas d_in[20+i]) fused on layers 0..5
  _Float16 *ci = actA, *co = actB;
  for (int l = 0; l < 7; ++l) {
    const float* alpha = (l < 6) ? (const float*)d_in[20 + l] : nullptr;
    conv3x3_wmma_kernel<<<cgrid, 256, 0, stream>>>(ci, co, wpk[l], bpk[l], alpha, H, W);
    _Float16* tmp = ci; ci = co; co = tmp;
  }
  // subnet output now in `ci` (== actB after 7 layers)

  blend_kernel<<<ew_blocks, 256, 0, stream>>>(pin, ci, wprm, xkbuf, (float*)d_out, B, H, W);
  build_b_kernel<<<ew_blocks, 256, 0, stream>>>(xkbuf, gy, actA, B, H, W);

  // stblock: 6 conv layers, PReLU(alphas d_in[38+i]) on layers 0..4
  ci = actA; co = actB;
  for (int l = 0; l < 6; ++l) {
    const float* alpha = (l < 5) ? (const float*)d_in[38 + l] : nullptr;
    conv3x3_wmma_kernel<<<cgrid, 256, 0, stream>>>(ci, co, wpk[7 + l], bpk[7 + l], alpha, H, W);
    _Float16* tmp = ci; ci = co; co = tmp;
  }
  // stblock output in `ci` (== actA after 6 layers)

  soft_thresh_kernel<<<ew_blocks, 256, 0, stream>>>(ci, dthr, (float*)d_out + nBHW, B, H, W);
}
